// Informer_47631187312755
// MI455X (gfx1250) — matrix operand order repaired and checked by reading, not compiled, not run on hardware
//
#include <hip/hip_runtime.h>
#include <math.h>

// ---------------------------------------------------------------------------
// Informer forward for MI455X (gfx1250).
// All matmuls go through v_wmma_f32_16x16x32_f16 (f16 operands, f32 accum).
// Wave32 shuffle reductions for softmax / layernorm / prob-attention measure.
// GEMM: LDS double-buffered staging (b128 global loads -> f16 b128 LDS stores)
// overlapped with WMMA compute; branch-free mask-multiply slow path for edge
// tiles / exotic strides; global_prefetch_b8 two K-tiles ahead.
// ---------------------------------------------------------------------------

typedef _Float16 v16h __attribute__((ext_vector_type(16)));
typedef _Float16 v8h  __attribute__((ext_vector_type(8)));
typedef float    v8f  __attribute__((ext_vector_type(8)));
typedef float    v4f  __attribute__((ext_vector_type(4)));

#define DM   512
#define NH   8
#define DHD  64
#define DFF  2048
#define BSZ  16
#define SEQ  720
#define DECL 672
#define PRED 336
#define LN_EPS 1e-5f
#define NEG_HUGE (-3.0e38f)

__device__ __forceinline__ float gelu_exact(float x) {
  return 0.5f * x * (1.0f + erff(x * 0.70710678118654752f));
}

__device__ __forceinline__ v8h cvt_f32x8_f16(v4f a, v4f b) {
  v8h h;
  h[0] = (_Float16)a[0]; h[1] = (_Float16)a[1];
  h[2] = (_Float16)a[2]; h[3] = (_Float16)a[3];
  h[4] = (_Float16)b[0]; h[5] = (_Float16)b[1];
  h[6] = (_Float16)b[2]; h[7] = (_Float16)b[3];
  return h;
}

// ------------------------------ WMMA GEMM ----------------------------------
// C[z] = act( A[z] @ B[z] + bias (+ C[z] if addC) )
// Block: 256 threads = 8 waves, block tile 64(M) x 64(N), K step 32.
// Waves: wm = wid>>1 (0..3) picks 16 rows, wn = wid&1 picks 32 cols (2 WMMAs).
// B staged transposed in LDS ([n][k]) so fragments are contiguous per lane.
// Double-buffered LDS tiles: stage tile t+1 while computing tile t.
#define TM 64
#define TN 64
#define TKK 32

__global__ __launch_bounds__(256)
void gemm_wmma_kernel(const float* __restrict__ A, const float* __restrict__ Bm,
                      const float* __restrict__ bias, float* __restrict__ C,
                      int M, int N, int K, int lda, int ldb, int ldbk, int ldc,
                      long long saB, long long saH, long long sbB, long long sbH,
                      long long scB, long long scH, int Hdiv,
                      int transB, int addC, int act)
{
  __shared__ _Float16 As[2][TM][TKK + 8];  // row pitch 40 halves = 80B (16B mult)
  __shared__ _Float16 Bs[2][TN][TKK + 8];

  int z  = blockIdx.z;
  int zb = z / Hdiv, zh = z - zb * Hdiv;
  const float* Ab = A + zb * saB + zh * saH;
  const float* Bb = Bm + zb * sbB + zh * sbH;
  float*       Cb = C + zb * scB + zh * scH;

  int m0 = blockIdx.y * TM;
  int n0 = blockIdx.x * TN;
  int tid  = threadIdx.x;
  int lane = tid & 31;
  int wid  = tid >> 5;
  int wm   = wid >> 1;
  int wn   = wid & 1;

  bool fullM = (m0 + TM <= M);
  bool fullN = (n0 + TN <= N);
  bool aligA = ((lda & 3) == 0);
  bool aligB = transB ? ((ldbk == 1) && ((ldb & 3) == 0)) : ((ldb & 3) == 0);

  // Stage one 64x32 A tile + 64x32 (transposed) B tile at k0 into buffer buf.
  auto stage = [&](int buf, int k0) {
    bool fullK = (k0 + TKK <= K);
    // ---------------- A tile ----------------
    if (fullM && fullK && aligA) {
      int r  = tid >> 2;
      int cb = (tid & 3) * 8;
      const float* ap = Ab + (long long)(m0 + r) * lda + (k0 + cb);
      if (k0 + 2 * TKK <= K) __builtin_prefetch(ap + TKK, 0, 1);
      v4f x0 = *(const v4f*)ap;
      v4f x1 = *(const v4f*)(ap + 4);
      *(v8h*)&As[buf][r][cb] = cvt_f32x8_f16(x0, x1);
    } else {
#pragma unroll
      for (int i = 0; i < 8; ++i) {
        int idx = tid + i * 256;
        int r = idx >> 5, c = idx & 31;
        int gm = m0 + r, gk = k0 + c;
        int cm = (gm < M) ? gm : (M - 1);
        int ck = (gk < K) ? gk : (K - 1);
        float v = Ab[(long long)cm * lda + ck];            // always in-bounds
        float msk = (gm < M && gk < K) ? 1.0f : 0.0f;      // forces the load
        As[buf][r][c] = (_Float16)(v * msk);
      }
    }
    // ------------- B tile (stored [n][k]) -------------
    if (fullN && fullK && aligB) {
      if (transB) {
        int n  = tid >> 2;
        int cb = (tid & 3) * 8;
        const float* bp = Bb + (long long)(n0 + n) * ldb + (k0 + cb);
        if (k0 + 2 * TKK <= K) __builtin_prefetch(bp + TKK, 0, 1);
        v4f x0 = *(const v4f*)bp;
        v4f x1 = *(const v4f*)(bp + 4);
        *(v8h*)&Bs[buf][n][cb] = cvt_f32x8_f16(x0, x1);
      } else {
        int kk = tid >> 3;
        int nb = (tid & 7) * 8;
        const float* bp = Bb + (long long)(k0 + kk) * ldb + (n0 + nb);
        if (k0 + 2 * TKK <= K) __builtin_prefetch(bp + (long long)TKK * ldb, 0, 1);
        v4f x0 = *(const v4f*)bp;
        v4f x1 = *(const v4f*)(bp + 4);
#pragma unroll
        for (int e = 0; e < 4; ++e) Bs[buf][nb + e][kk]     = (_Float16)x0[e];
#pragma unroll
        for (int e = 0; e < 4; ++e) Bs[buf][nb + 4 + e][kk] = (_Float16)x1[e];
      }
    } else {
#pragma unroll
      for (int i = 0; i < 8; ++i) {
        int idx = tid + i * 256;
        int n = idx >> 5, c = idx & 31;
        int gn = n0 + n, gk = k0 + c;
        int cn = (gn < N) ? gn : (N - 1);
        int ck = (gk < K) ? gk : (K - 1);
        long long bi = transB ? ((long long)cn * ldb + (long long)ck * ldbk)
                              : ((long long)ck * ldb + cn);
        float v = Bb[bi];                                   // always in-bounds
        float msk = (gn < N && gk < K) ? 1.0f : 0.0f;
        Bs[buf][n][c] = (_Float16)(v * msk);
      }
    }
  };

  v8f acc0 = {};
  v8f acc1 = {};

  int nk = (K + TKK - 1) / TKK;
  stage(0, 0);
  for (int t = 0; t < nk; ++t) {
    __syncthreads();                 // tile t visible; prior reads of buf done
    if (t + 1 < nk) stage((t + 1) & 1, (t + 1) * TKK);
    int buf = t & 1;

    // A fragment (16x32 f16): lane L<16 holds row, K 0..7 & 16..23 ; L>=16: +8
    int arow = wm * 16 + (lane & 15);
    int akb  = (lane >> 4) * 8;
    v16h af;
#pragma unroll
    for (int e = 0; e < 8; ++e) {
      af[e]     = As[buf][arow][akb + e];
      af[e + 8] = As[buf][arow][akb + 16 + e];
    }
    // B fragments (32x16 f16): lanes 0-15 K=0..7 & 16..23, lanes 16-31: +8
    int bkb = (lane >> 4) * 8;
    int bn0 = wn * 32 + (lane & 15);
    v16h bf0, bf1;
#pragma unroll
    for (int e = 0; e < 8; ++e) {
      bf0[e]     = Bs[buf][bn0][bkb + e];
      bf0[e + 8] = Bs[buf][bn0][bkb + 16 + e];
      bf1[e]     = Bs[buf][bn0 + 16][bkb + e];
      bf1[e + 8] = Bs[buf][bn0 + 16][bkb + 16 + e];
    }
    acc0 = __builtin_amdgcn_wmma_f32_16x16x32_f16(false, af, false, bf0,
                                                  (short)0, acc0, false, false);
    acc1 = __builtin_amdgcn_wmma_f32_16x16x32_f16(false, af, false, bf1,
                                                  (short)0, acc1, false, false);
  }

  // ------------------------------ epilogue ---------------------------------
  // D layout: lane<16 rows 0..7 in vgpr 0..7; lane>=16 rows 8..15
  int crow0 = m0 + wm * 16 + (lane >> 4) * 8;
  if (fullM && fullN) {
#pragma unroll
    for (int t = 0; t < 2; ++t) {
      int ccol = n0 + wn * 32 + t * 16 + (lane & 15);
      float bv = bias ? bias[ccol] : 0.0f;
#pragma unroll
      for (int r = 0; r < 8; ++r) {
        int crow = crow0 + r;
        float v = (t ? acc1[r] : acc0[r]) + bv;
        long long ci = (long long)crow * ldc + ccol;
        if (addC) v += Cb[ci];
        if (act == 1) v = gelu_exact(v);
        Cb[ci] = v;
      }
    }
  } else {
#pragma unroll
    for (int t = 0; t < 2; ++t) {
      int ccol = n0 + wn * 32 + t * 16 + (lane & 15);
      if (ccol >= N) continue;
      float bv = bias ? bias[ccol] : 0.0f;
#pragma unroll
      for (int r = 0; r < 8; ++r) {
        int crow = crow0 + r;
        if (crow >= M) continue;
        float v = (t ? acc1[r] : acc0[r]) + bv;
        long long ci = (long long)crow * ldc + ccol;
        if (addC) v += Cb[ci];
        if (act == 1) v = gelu_exact(v);
        Cb[ci] = v;
      }
    }
  }
}

// ------------------------------ Embedding ----------------------------------
// out[b][l][d] = circ_conv(x, convw) + sinusoid(l,d) + sum_m sinusoid(mark,d)
__global__ void embed_kernel(const float* __restrict__ x,
                             const int* __restrict__ mark,
                             const float* __restrict__ convw,
                             float* __restrict__ out, int L, int Cin)
{
  long long idx = (long long)blockIdx.x * blockDim.x + threadIdx.x;
  long long total = (long long)BSZ * L * DM;
  if (idx >= total) return;
  int d = (int)(idx % DM);
  int l = (int)((idx / DM) % L);
  int b = (int)(idx / ((long long)DM * L));

  float acc = 0.0f;
#pragma unroll
  for (int t = 0; t < 3; ++t) {
    int ls = l - 1 + t; ls += (ls < 0) ? L : 0; ls -= (ls >= L) ? L : 0;
    const float* xp = x + ((long long)b * L + ls) * Cin;
    for (int i = 0; i < Cin; ++i)
      acc += xp[i] * convw[(d * Cin + i) * 3 + t];
  }
  int j = d >> 1;
  float freq = __expf(-(float)(2 * j) * 9.21034037e-3f); // ln(1e4)/1000 * 2j
  float ang = (float)l * freq;
  acc += (d & 1) ? __cosf(ang) : __sinf(ang);
  const int* mk = mark + ((long long)b * L + l) * 4;
#pragma unroll
  for (int m = 0; m < 4; ++m) {
    float a2 = (float)mk[m] * freq;
    acc += (d & 1) ? __cosf(a2) : __sinf(a2);
  }
  out[idx] = acc;
}

// ------------------------- Fused residual + LayerNorm ----------------------
// One wave per row of 512; 16 elements/lane held in registers.
__global__ void layernorm_kernel(const float* __restrict__ X,
                                 const float* __restrict__ R,
                                 const float* __restrict__ g,
                                 const float* __restrict__ bb,
                                 float* __restrict__ O, int rows)
{
  int wid = threadIdx.x >> 5, lane = threadIdx.x & 31;
  int row = blockIdx.x * 8 + wid;
  if (row >= rows) return;
  const float* xp = X + (long long)row * DM;
  const float* rp = R ? R + (long long)row * DM : nullptr;
  float vbuf[16];
  float s = 0.0f, s2 = 0.0f;
#pragma unroll
  for (int i = 0; i < 16; ++i) {
    int c = lane + 32 * i;
    float v = xp[c] + (rp ? rp[c] : 0.0f);
    vbuf[i] = v; s += v; s2 += v * v;
  }
#pragma unroll
  for (int o = 16; o > 0; o >>= 1) {
    s  += __shfl_xor(s, o, 32);
    s2 += __shfl_xor(s2, o, 32);
  }
  float mean = s * (1.0f / DM);
  float var  = s2 * (1.0f / DM) - mean * mean;
  float rs   = rsqrtf(var + LN_EPS);
  float* op = O + (long long)row * DM;
#pragma unroll
  for (int i = 0; i < 16; ++i) {
    int c = lane + 32 * i;
    op[c] = g[c] * (vbuf[i] - mean) * rs + bb[c];
  }
}

// ------------------------------ Softmax ------------------------------------
// One wave per row; optional causal mask (col > q masked), scale pre-softmax.
__global__ void softmax_kernel(float* __restrict__ S, int rows, int cols,
                               float scale, int causal, int Lq)
{
  int wid = threadIdx.x >> 5, lane = threadIdx.x & 31;
  int row = blockIdx.x * 8 + wid;
  if (row >= rows) return;
  float* sp = S + (long long)row * cols;
  int q = row % Lq;
  int lim = causal ? (q + 1) : cols;
  float mx = NEG_HUGE;
  for (int c = lane; c < lim; c += 32) mx = fmaxf(mx, sp[c] * scale);
#pragma unroll
  for (int o = 16; o > 0; o >>= 1) mx = fmaxf(mx, __shfl_xor(mx, o, 32));
  float sm = 0.0f;
  for (int c = lane; c < lim; c += 32) sm += __expf(sp[c] * scale - mx);
#pragma unroll
  for (int o = 16; o > 0; o >>= 1) sm += __shfl_xor(sm, o, 32);
  float inv = 1.0f / sm;
  for (int c = lane; c < cols; c += 32)
    sp[c] = (c < lim) ? __expf(sp[c] * scale - mx) * inv : 0.0f;
}

// -------------------- ProbSparse attention helpers -------------------------
// Sparsity measure m[q] = max_s(q.k_s) - mean_s(q.k_s) over hashed key samples
__global__ void prob_measure_kernel(const float* __restrict__ Q,
                                    const float* __restrict__ K,
                                    float* __restrict__ Mout,
                                    int Lq, int Lk, int upart)
{
  int gid  = blockIdx.x * 8 + (threadIdx.x >> 5);
  int lane = threadIdx.x & 31;
  int total = BSZ * NH * Lq;
  if (gid >= total) return;
  int q = gid % Lq;
  int h = (gid / Lq) % NH;
  int b = gid / (Lq * NH);
  const float* qv = Q + ((long long)(b * Lq + q)) * DM + h * DHD;
  float mx = NEG_HUGE, sm = 0.0f;
  for (int s = 0; s < upart; ++s) {
    unsigned hsh = (unsigned)(q * 131 + s) * 2654435761u + 12345u;
    int kidx = (int)(hsh % (unsigned)Lk);
    const float* kv = K + ((long long)(b * Lk + kidx)) * DM + h * DHD;
    float p = qv[2 * lane] * kv[2 * lane] + qv[2 * lane + 1] * kv[2 * lane + 1];
#pragma unroll
    for (int o = 16; o > 0; o >>= 1) p += __shfl_xor(p, o, 32);
    mx = fmaxf(mx, p); sm += p;
  }
  if (lane == 0) Mout[gid] = mx - sm / (float)upart;
}

// Per (b,h): iterative block argmax to select top-u query indices
__global__ void topk_kernel(const float* __restrict__ Mv, int* __restrict__ top,
                            int Lq, int u)
{
  __shared__ float vals[768];
  __shared__ float rv[256];
  __shared__ int   ri[256];
  int z = blockIdx.x;
  const float* mrow = Mv + (long long)z * Lq;
  for (int i = threadIdx.x; i < Lq; i += 256) vals[i] = mrow[i];
  __syncthreads();
  for (int j = 0; j < u; ++j) {
    float best = NEG_HUGE; int bi = 0;
    for (int i = threadIdx.x; i < Lq; i += 256)
      if (vals[i] > best) { best = vals[i]; bi = i; }
    rv[threadIdx.x] = best; ri[threadIdx.x] = bi;
    __syncthreads();
    for (int sft = 128; sft > 0; sft >>= 1) {
      if (threadIdx.x < sft && rv[threadIdx.x + sft] > rv[threadIdx.x]) {
        rv[threadIdx.x] = rv[threadIdx.x + sft];
        ri[threadIdx.x] = ri[threadIdx.x + sft];
      }
      __syncthreads();
    }
    if (threadIdx.x == 0) { top[z * u + j] = ri[0]; vals[ri[0]] = NEG_HUGE; }
    __syncthreads();
  }
}

__global__ void gather_q_kernel(const float* __restrict__ Q,
                                const int* __restrict__ top,
                                float* __restrict__ Qred, int Lq, int u)
{
  int idx = blockIdx.x * blockDim.x + threadIdx.x;
  int total = BSZ * NH * u * DHD;
  if (idx >= total) return;
  int d = idx & 63;
  int j = (idx >> 6) % u;
  int z = idx / (64 * u);
  int h = z % NH, b = z / NH;
  int q = top[z * u + j];
  Qred[idx] = Q[((long long)(b * Lq + q)) * DM + h * DHD + d];
}

__global__ void vmean_kernel(const float* __restrict__ V,
                             float* __restrict__ vm, int Lk)
{
  int idx = blockIdx.x * blockDim.x + threadIdx.x;
  if (idx >= BSZ * NH * DHD) return;
  int d = idx & 63;
  int h = (idx >> 6) % NH;
  int b = idx / (64 * NH);
  const float* vp = V + (long long)b * Lk * DM + h * DHD + d;
  float s = 0.0f;
  for (int k = 0; k < Lk; ++k) s += vp[(long long)k * DM];
  vm[idx] = s / (float)Lk;
}

__global__ void ctx_fill_kernel(const float* __restrict__ vm,
                                float* __restrict__ ctx, int Lq)
{
  long long idx = (long long)blockIdx.x * blockDim.x + threadIdx.x;
  long long total = (long long)BSZ * Lq * DM;
  if (idx >= total) return;
  int d512 = (int)(idx % DM);
  int h = d512 >> 6, d = d512 & 63;
  long long bl = idx / DM;
  int b = (int)(bl / Lq);
  ctx[idx] = vm[((long long)b * NH + h) * DHD + d];
}

__global__ void ctx_scatter_kernel(const float* __restrict__ out_top,
                                   const int* __restrict__ top,
                                   float* __restrict__ ctx, int Lq, int u)
{
  int idx = blockIdx.x * blockDim.x + threadIdx.x;
  int total = BSZ * NH * u * DHD;
  if (idx >= total) return;
  int d = idx & 63;
  int j = (idx >> 6) % u;
  int z = idx / (64 * u);
  int h = z % NH, b = z / NH;
  int q = top[z * u + j];
  ctx[((long long)(b * Lq + q)) * DM + h * DHD + d] = out_top[idx];
}

// ------------------------- Distil (conv/bn/elu/pool) -----------------------
__global__ void pad_wrap_kernel(const float* __restrict__ X,
                                float* __restrict__ XP, int L)
{
  long long idx = (long long)blockIdx.x * blockDim.x + threadIdx.x;
  long long total = (long long)BSZ * (L + 2) * DM;
  if (idx >= total) return;
  int c = (int)(idx % DM);
  int lp = (int)((idx / DM) % (L + 2));
  int b = (int)(idx / ((long long)DM * (L + 2)));
  int ls = lp - 1; ls += (ls < 0) ? L : 0; ls -= (ls >= L) ? L : 0;
  XP[idx] = X[((long long)(b * L + ls)) * DM + c];
}

__global__ void bn_stats_kernel(const float* __restrict__ Y,
                                float* __restrict__ mean,
                                float* __restrict__ var, int rows)
{
  __shared__ float s1[256], s2[256];
  int c = blockIdx.x;
  float a = 0.0f, b2 = 0.0f;
  for (int r = threadIdx.x; r < rows; r += 256) {
    float v = Y[(long long)r * DM + c];
    a += v; b2 += v * v;
  }
  s1[threadIdx.x] = a; s2[threadIdx.x] = b2;
  __syncthreads();
  for (int sft = 128; sft > 0; sft >>= 1) {
    if (threadIdx.x < sft) {
      s1[threadIdx.x] += s1[threadIdx.x + sft];
      s2[threadIdx.x] += s2[threadIdx.x + sft];
    }
    __syncthreads();
  }
  if (threadIdx.x == 0) {
    float m = s1[0] / (float)rows;
    mean[c] = m;
    var[c]  = s2[0] / (float)rows - m * m;
  }
}

__global__ void bn_apply_elu_kernel(float* __restrict__ Y,
                                    const float* __restrict__ mean,
                                    const float* __restrict__ var,
                                    const float* __restrict__ g,
                                    const float* __restrict__ bb, int rows)
{
  long long idx = (long long)blockIdx.x * blockDim.x + threadIdx.x;
  long long total = (long long)rows * DM;
  if (idx >= total) return;
  int c = (int)(idx % DM);
  float v = Y[idx];
  v = g[c] * (v - mean[c]) * rsqrtf(var[c] + LN_EPS) + bb[c];
  Y[idx] = (v > 0.0f) ? v : expm1f(v);
}

__global__ void maxpool_kernel(const float* __restrict__ Y,
                               float* __restrict__ O, int L, int Lout)
{
  long long idx = (long long)blockIdx.x * blockDim.x + threadIdx.x;
  long long total = (long long)BSZ * Lout * DM;
  if (idx >= total) return;
  int c = (int)(idx % DM);
  int lo = (int)((idx / DM) % Lout);
  int b = (int)(idx / ((long long)DM * Lout));
  int li0 = 2 * lo - 1;
  float m = NEG_HUGE;
#pragma unroll
  for (int t = 0; t < 3; ++t) {
    int li = li0 + t;
    if (li >= 0 && li < L) m = fmaxf(m, Y[((long long)(b * L + li)) * DM + c]);
  }
  O[idx] = m;
}

// ------------------------------ Output slice -------------------------------
__global__ void slice_out_kernel(const float* __restrict__ full,
                                 float* __restrict__ out)
{
  int idx = blockIdx.x * blockDim.x + threadIdx.x;
  int total = BSZ * PRED * 7;
  if (idx >= total) return;
  int c = idx % 7;
  int l = (idx / 7) % PRED;
  int b = idx / (7 * PRED);
  out[idx] = full[((long long)(b * DECL + (DECL - PRED) + l)) * 7 + c];
}

// ===========================================================================
//                                Host side
// ===========================================================================
static inline int ceildiv(int a, int b) { return (a + b - 1) / b; }

static void launch_gemm(hipStream_t st, const float* A, const float* Bm,
                        const float* bias, float* C, int M, int N, int K,
                        int lda, int ldb, int ldbk, int ldc,
                        int batch, int Hdiv,
                        long long saB, long long saH,
                        long long sbB, long long sbH,
                        long long scB, long long scH,
                        int transB, int addC, int act)
{
  dim3 g(ceildiv(N, TN), ceildiv(M, TM), batch);
  gemm_wmma_kernel<<<g, dim3(256), 0, st>>>(A, Bm, bias, C, M, N, K,
                                            lda, ldb, ldbk, ldc,
                                            saB, saH, sbB, sbH, scB, scH,
                                            Hdiv, transB, addC, act);
}

extern "C" void kernel_launch(void* const* d_in, const int* in_sizes, int n_in,
                              void* d_out, int out_size, void* d_ws,
                              size_t ws_size, hipStream_t stream)
{
  (void)in_sizes; (void)n_in; (void)out_size; (void)ws_size;
  const float* x_enc   = (const float*)d_in[0];
  const int*   mk_enc  = (const int*)d_in[1];
  const float* x_dec   = (const float*)d_in[2];
  const int*   mk_dec  = (const int*)d_in[3];
  // Params: JAX pytree order (alphabetical keys at each dict level).
  const float* P[58];
  for (int i = 0; i < 58; ++i) P[i] = (const float*)d_in[4 + i];
  // 0 c0.b 1 c0.bn_b 2 c0.bn_g 3 c0.w 4 c1.b 5 c1.bn_b 6 c1.bn_g 7 c1.w
  // 8 dec_emb_conv 9 d.W1 10 d.W2 11 d.b1 12 d.b2
  // 13 d.cr.Wk 14 d.cr.Wo 15 d.cr.Wq 16 d.cr.Wv 17 d.cr.bk 18 d.cr.bo 19 d.cr.bq 20 d.cr.bv
  // 21 d.ln1_b 22 d.ln1_g 23 d.ln2_b 24 d.ln2_g 25 d.ln3_b 26 d.ln3_g
  // 27 d.sf.Wk 28 d.sf.Wo 29 d.sf.Wq 30 d.sf.Wv 31 d.sf.bk 32 d.sf.bo 33 d.sf.bq 34 d.sf.bv
  // 35 dec_norm_b 36 dec_norm_g 37 enc_emb_conv
  // 38 e.W1 39 e.W2 40 e.Wk 41 e.Wo 42 e.Wq 43 e.Wv 44 e.b1 45 e.b2
  // 46 e.bk 47 e.bo 48 e.bq 49 e.bv 50 e.ln1_b 51 e.ln1_g 52 e.ln2_b 53 e.ln2_g
  // 54 enc_norm_b 55 enc_norm_g 56 proj_W 57 proj_b

  float* W = (float*)d_ws;
  size_t off = 0;
  auto alloc = [&](size_t n) -> float* {
    float* p = W + off;
    off += (n + 63) & ~(size_t)63;
    return p;
  };
  float* Xa  = alloc((size_t)BSZ * SEQ * DM);
  float* Xb  = alloc((size_t)BSZ * SEQ * DM);
  float* Qb  = alloc((size_t)BSZ * SEQ * DM);
  float* Kb  = alloc((size_t)BSZ * SEQ * DM);
  float* Vb  = alloc((size_t)BSZ * SEQ * DM);
  float* CT  = alloc((size_t)BSZ * SEQ * DM);
  float* AO  = alloc((size_t)BSZ * SEQ * DM);
  float* FH  = alloc((size_t)BSZ * SEQ * DFF);
  float* FO  = alloc((size_t)BSZ * SEQ * DM);
  float* S   = alloc((size_t)BSZ * NH * DECL * DECL);
  float* ENC = alloc((size_t)BSZ * 180 * DM);
  float* XP  = alloc((size_t)BSZ * (SEQ + 2) * DM);
  float* Mv  = alloc((size_t)BSZ * NH * SEQ);
  float* VM  = alloc((size_t)BSZ * NH * DHD);
  float* QR  = alloc((size_t)BSZ * NH * 64 * DHD);
  float* OT  = alloc((size_t)BSZ * NH * 64 * DHD);
  int*   TOPI = (int*)alloc((size_t)BSZ * NH * 64);
  float* BNM = alloc(DM);
  float* BNV = alloc(DM);
  float* FULL = alloc((size_t)BSZ * DECL * 7);

  // =========================== Encoder ===========================
  int L = SEQ;
  embed_kernel<<<ceildiv(BSZ * L * DM, 256), 256, 0, stream>>>(
      x_enc, mk_enc, P[37], Xa, L, 7);
  float* cur = Xa; float* oth = Xb;

  for (int layer = 0; layer < 3; ++layer) {
    int rows = BSZ * L;
    // Q,K,V projections
    launch_gemm(stream, cur, P[42], P[48], Qb, rows, DM, DM, DM, DM, 1, DM,
                1, 1, 0, 0, 0, 0, 0, 0, 0, 0, 0);
    launch_gemm(stream, cur, P[40], P[46], Kb, rows, DM, DM, DM, DM, 1, DM,
                1, 1, 0, 0, 0, 0, 0, 0, 0, 0, 0);
    launch_gemm(stream, cur, P[43], P[49], Vb, rows, DM, DM, DM, DM, 1, DM,
                1, 1, 0, 0, 0, 0, 0, 0, 0, 0, 0);
    // ProbSparse attention
    int u = 5 * (int)ceilf(logf((float)L));
    if (u > L) u = L;
    prob_measure_kernel<<<ceildiv(BSZ * NH * L, 8), 256, 0, stream>>>(
        Qb, Kb, Mv, L, L, u);
    topk_kernel<<<BSZ * NH, 256, 0, stream>>>(Mv, TOPI, L, u);
    gather_q_kernel<<<ceildiv(BSZ * NH * u * DHD, 256), 256, 0, stream>>>(
        Qb, TOPI, QR, L, u);
    // scores (u x L) per (b,h): Qred @ K^T
    launch_gemm(stream, QR, Kb, nullptr, S, u, L, DHD,
                DHD, DM, 1, L, BSZ * NH, NH,
                (long long)NH * u * DHD, (long long)u * DHD,
                (long long)L * DM, DHD,
                (long long)NH * u * L, (long long)u * L, 1, 0, 0);
    softmax_kernel<<<ceildiv(BSZ * NH * u, 8), 256, 0, stream>>>(
        S, BSZ * NH * u, L, 0.125f, 0, u);
    // out_top = P @ V
    launch_gemm(stream, S, Vb, nullptr, OT, u, DHD, L,
                L, DM, 1, DHD, BSZ * NH, NH,
                (long long)NH * u * L, (long long)u * L,
                (long long)L * DM, DHD,
                (long long)NH * u * DHD, (long long)u * DHD, 0, 0, 0);
    vmean_kernel<<<ceildiv(BSZ * NH * DHD, 256), 256, 0, stream>>>(Vb, VM, L);
    ctx_fill_kernel<<<ceildiv(BSZ * L * DM, 256), 256, 0, stream>>>(VM, CT, L);
    ctx_scatter_kernel<<<ceildiv(BSZ * NH * u * DHD, 256), 256, 0, stream>>>(
        OT, TOPI, CT, L, u);
    // output projection + residual LN
    launch_gemm(stream, CT, P[41], P[47], AO, rows, DM, DM, DM, DM, 1, DM,
                1, 1, 0, 0, 0, 0, 0, 0, 0, 0, 0);
    layernorm_kernel<<<ceildiv(rows, 8), 256, 0, stream>>>(
        cur, AO, P[51], P[50], oth, rows);
    // FFN (GELU fused into GEMM epilogue)
    launch_gemm(stream, oth, P[38], P[44], FH, rows, DFF, DM, DM, DFF, 1, DFF,
                1, 1, 0, 0, 0, 0, 0, 0, 0, 0, 1);
    launch_gemm(stream, FH, P[39], P[45], FO, rows, DM, DFF, DFF, DM, 1, DM,
                1, 1, 0, 0, 0, 0, 0, 0, 0, 0, 0);
    layernorm_kernel<<<ceildiv(rows, 8), 256, 0, stream>>>(
        oth, FO, P[53], P[52], cur, rows);

    if (layer < 2) {  // distil: circular conv (3 GEMM taps) + BN + ELU + pool
      const float* cw = (layer == 0) ? P[3] : P[7];
      const float* cb = (layer == 0) ? P[0] : P[4];
      const float* bg = (layer == 0) ? P[2] : P[6];
      const float* bB = (layer == 0) ? P[1] : P[5];
      pad_wrap_kernel<<<ceildiv(BSZ * (L + 2) * DM, 256), 256, 0, stream>>>(
          cur, XP, L);
      for (int t = 0; t < 3; ++t) {
        launch_gemm(stream, XP + (size_t)t * DM, cw + t,
                    (t == 0) ? cb : nullptr, FO, L, DM, DM,
                    DM, 3 * DM, 3, DM, BSZ, 1,
                    (long long)(L + 2) * DM, 0, 0, 0,
                    (long long)L * DM, 0, 1, (t > 0) ? 1 : 0, 0);
      }
      bn_stats_kernel<<<DM, 256, 0, stream>>>(FO, BNM, BNV, BSZ * L);
      bn_apply_elu_kernel<<<ceildiv(BSZ * L * DM, 256), 256, 0, stream>>>(
          FO, BNM, BNV, bg, bB, BSZ * L);
      int Lout = (L - 1) / 2 + 1;
      maxpool_kernel<<<ceildiv(BSZ * Lout * DM, 256), 256, 0, stream>>>(
          FO, cur, L, Lout);
      L = Lout;
    }
  }
  int Lenc = L;  // 180
  layernorm_kernel<<<ceildiv(BSZ * Lenc, 8), 256, 0, stream>>>(
      cur, nullptr, P[55], P[54], ENC, BSZ * Lenc);

  // =========================== Decoder ===========================
  embed_kernel<<<ceildiv(BSZ * DECL * DM, 256), 256, 0, stream>>>(
      x_dec, mk_dec, P[8], Xa, DECL, 7);
  float* dc = Xa; float* dn = Xb;
  int drows = BSZ * DECL;

  for (int layer = 0; layer < 2; ++layer) {
    // ---- masked self attention ----
    launch_gemm(stream, dc, P[29], P[33], Qb, drows, DM, DM, DM, DM, 1, DM,
                1, 1, 0, 0, 0, 0, 0, 0, 0, 0, 0);
    launch_gemm(stream, dc, P[27], P[31], Kb, drows, DM, DM, DM, DM, 1, DM,
                1, 1, 0, 0, 0, 0, 0, 0, 0, 0, 0);
    launch_gemm(stream, dc, P[30], P[34], Vb, drows, DM, DM, DM, DM, 1, DM,
                1, 1, 0, 0, 0, 0, 0, 0, 0, 0, 0);
    launch_gemm(stream, Qb, Kb, nullptr, S, DECL, DECL, DHD,
                DM, DM, 1, DECL, BSZ * NH, NH,
                (long long)DECL * DM, DHD,
                (long long)DECL * DM, DHD,
                (long long)NH * DECL * DECL, (long long)DECL * DECL, 1, 0, 0);
    softmax_kernel<<<ceildiv(BSZ * NH * DECL, 8), 256, 0, stream>>>(
        S, BSZ * NH * DECL, DECL, 0.125f, 1, DECL);
    launch_gemm(stream, S, Vb, nullptr, CT, DECL, DHD, DECL,
                DECL, DM, 1, DM, BSZ * NH, NH,
                (long long)NH * DECL * DECL, (long long)DECL * DECL,
                (long long)DECL * DM, DHD,
                (long long)DECL * DM, DHD, 0, 0, 0);
    launch_gemm(stream, CT, P[28], P[32], AO, drows, DM, DM, DM, DM, 1, DM,
                1, 1, 0, 0, 0, 0, 0, 0, 0, 0, 0);
    layernorm_kernel<<<ceildiv(drows, 8), 256, 0, stream>>>(
        dc, AO, P[22], P[21], dn, drows);
    { float* t = dc; dc = dn; dn = t; }
    // ---- cross attention ----
    launch_gemm(stream, dc, P[15], P[19], Qb, drows, DM, DM, DM, DM, 1, DM,
                1, 1, 0, 0, 0, 0, 0, 0, 0, 0, 0);
    launch_gemm(stream, ENC, P[13], P[17], Kb, BSZ * Lenc, DM, DM, DM, DM, 1,
                DM, 1, 1, 0, 0, 0, 0, 0, 0, 0, 0, 0);
    launch_gemm(stream, ENC, P[16], P[20], Vb, BSZ * Lenc, DM, DM, DM, DM, 1,
                DM, 1, 1, 0, 0, 0, 0, 0, 0, 0, 0, 0);
    launch_gemm(stream, Qb, Kb, nullptr, S, DECL, Lenc, DHD,
                DM, DM, 1, Lenc, BSZ * NH, NH,
                (long long)DECL * DM, DHD,
                (long long)Lenc * DM, DHD,
                (long long)NH * DECL * Lenc, (long long)DECL * Lenc, 1, 0, 0);
    softmax_kernel<<<ceildiv(BSZ * NH * DECL, 8), 256, 0, stream>>>(
        S, BSZ * NH * DECL, Lenc, 0.125f, 0, DECL);
    launch_gemm(stream, S, Vb, nullptr, CT, DECL, DHD, Lenc,
                Lenc, DM, 1, DM, BSZ * NH, NH,
                (long long)NH * DECL * Lenc, (long long)DECL * Lenc,
                (long long)Lenc * DM, DHD,
                (long long)DECL * DM, DHD, 0, 0, 0);
    launch_gemm(stream, CT, P[14], P[18], AO, drows, DM, DM, DM, DM, 1, DM,
                1, 1, 0, 0, 0, 0, 0, 0, 0, 0, 0);
    layernorm_kernel<<<ceildiv(drows, 8), 256, 0, stream>>>(
        dc, AO, P[24], P[23], dn, drows);
    { float* t = dc; dc = dn; dn = t; }
    // ---- FFN ----
    launch_gemm(stream, dc, P[9], P[11], FH, drows, DFF, DM, DM, DFF, 1, DFF,
                1, 1, 0, 0, 0, 0, 0, 0, 0, 0, 1);
    launch_gemm(stream, FH, P[10], P[12], FO, drows, DM, DFF, DFF, DM, 1, DM,
                1, 1, 0, 0, 0, 0, 0, 0, 0, 0, 0);
    layernorm_kernel<<<ceildiv(drows, 8), 256, 0, stream>>>(
        dc, FO, P[26], P[25], dn, drows);
    { float* t = dc; dc = dn; dn = t; }
  }
  layernorm_kernel<<<ceildiv(drows, 8), 256, 0, stream>>>(
      dc, nullptr, P[36], P[35], dn, drows);
  // Final projection to 7 channels + slice last PRED steps
  launch_gemm(stream, dn, P[56], P[57], FULL, drows, 7, DM, DM, 7, 1, 7,
              1, 1, 0, 0, 0, 0, 0, 0, 0, 0, 0);
  slice_out_kernel<<<ceildiv(BSZ * PRED * 7, 256), 256, 0, stream>>>(
      FULL, (float*)d_out);
}
